// UVUTensorProduct_17420387352803
// MI455X (gfx1250) — matrix operand order repaired and checked by reading, not compiled, not run on hardware
//
#include <hip/hip_runtime.h>
#include <hip/hip_bf16.h>

typedef __attribute__((ext_vector_type(16))) _Float16 v16h;
typedef __attribute__((ext_vector_type(8)))  float    v8f;

#define ROWS_PER_BLOCK 96           // 6 waves x 16 rows
#define THREADS        192          // 6 wave32 waves
#define LOG2_F         0.6931471805599453f
#define SSP_CST_F      1.8755968f   // 1/sqrt(E[(softplus(x)-log2)^2]), x~N(0,1)
#define INV_SQRT3_F    0.5773502691896258f
#define INV_SQRT8_F    0.35355339059327373f

__global__ __launch_bounds__(THREADS)
void UVUTensorProduct_17420387352803_kernel(const float* __restrict__ data1,
                                            const float* __restrict__ data2,
                                            const float* __restrict__ dweight,
                                            const float* __restrict__ w0,
                                            const float* __restrict__ w1,
                                            float* __restrict__ out)
{
    // LDS: w1 pre-swizzled into WMMA-B fragment layout [ntile][kchunk][lane][half]
    __shared__ __align__(32) _Float16 lds_w1[16 * 2 * 32 * 16];   // 32 KB
    __shared__ __align__(32) _Float16 lds_h[ROWS_PER_BLOCK * 64]; // 12 KB
    __shared__ float lds_w0[8 * 64];                              //  2 KB
    __shared__ float lds_dw[ROWS_PER_BLOCK * 8];                  //  3 KB
    __shared__ float lds_x2[ROWS_PER_BLOCK * 4];                  //1.5 KB

    const int t      = threadIdx.x;
    const int eblock = blockIdx.x * ROWS_PER_BLOCK;

    // ---- Phase 0: stage weights + per-row small inputs; prefetch data1 tile ----
    for (int d = t * 64; d < ROWS_PER_BLOCK * 256; d += THREADS * 64)
        __builtin_prefetch(&data1[(size_t)eblock * 256 + d], 0, 3);

    // w1/8 -> f16, swizzled to B-fragment order:
    //   k = kchunk*32 + (lane>=16)*16 + half ;  n = ntile*16 + (lane&15)
    for (int d = t; d < 16384; d += THREADS) {
        int ntile  = d >> 10;
        int kchunk = (d >> 9) & 1;
        int lane   = (d >> 4) & 31;
        int hh     = d & 15;
        int k = kchunk * 32 + (lane >> 4) * 16 + hh;
        int n = ntile * 16 + (lane & 15);
        lds_w1[d] = (_Float16)(w1[k * 256 + n] * 0.125f);
    }
    for (int d = t; d < 512; d += THREADS)
        lds_w0[d] = w0[d] * INV_SQRT8_F;
    for (int d = t; d < ROWS_PER_BLOCK * 8; d += THREADS)
        lds_dw[d] = dweight[(size_t)eblock * 8 + d];
    for (int d = t; d < ROWS_PER_BLOCK * 4; d += THREADS)
        lds_x2[d] = data2[(size_t)eblock * 4 + d];
    __syncthreads();

    // ---- Phase 1: h = ssp(dw @ w0s), 96x64, two threads per row ----
    {
        const int erow = t >> 1;
        const int j0   = (t & 1) * 32;
        float dwv[8];
        #pragma unroll
        for (int k = 0; k < 8; ++k) dwv[k] = lds_dw[erow * 8 + k];
        for (int j = j0; j < j0 + 32; ++j) {
            float g = 0.f;
            #pragma unroll
            for (int k = 0; k < 8; ++k) g = fmaf(dwv[k], lds_w0[k * 64 + j], g);
            float sp = fmaxf(g, 0.f) + log1pf(expf(-fabsf(g)));   // stable softplus
            lds_h[erow * 64 + j] = (_Float16)((sp - LOG2_F) * SSP_CST_F);
        }
    }
    __syncthreads();

    // ---- Phase 2: per-wave 16-row GEMM tile via WMMA + fused epilogue ----
    const int wave  = t >> 5;
    const int lane  = t & 31;
    const int erow0 = wave * 16;      // local row base of this wave's tile
    const int m     = lane & 15;      // A row index / D column index
    const int hi    = lane >> 4;      // lane group

    // A fragments (16x32 f16, K-chunks 0 and 1) from lds_h.
    // Layout: lanes 0-15 hold K {0..15}; lanes 16-31 hold K {16..31}.
    v16h a0, a1;
    {
        const _Float16* hrow = &lds_h[(erow0 + m) * 64];
        const int kb = hi * 16;
        #pragma unroll
        for (int i = 0; i < 8; ++i) {
            a0[i]     = hrow[kb + i];
            a0[i + 8] = hrow[kb + 8 + i];
            a1[i]     = hrow[32 + kb + i];
            a1[i + 8] = hrow[32 + kb + 8 + i];
        }
    }

    // Full unroll: ntile (and j = ntile>>2) become compile-time constants, so
    // the 4-way epilogue select folds to straight-line code per tile.
    #pragma unroll
    for (int ntile = 0; ntile < 16; ++ntile) {
        const _Float16* bbase = &lds_w1[ntile * 1024];
        v16h b0 = *(const v16h*)(bbase + lane * 16);         // kchunk 0
        v16h b1 = *(const v16h*)(bbase + 512 + lane * 16);   // kchunk 1

        v8f acc = {};
        acc = __builtin_amdgcn_wmma_f32_16x16x32_f16(false, a0, false, b0,
                                                     (short)0, acc, false, false);
        acc = __builtin_amdgcn_wmma_f32_16x16x32_f16(false, a1, false, b1,
                                                     (short)0, acc, false, false);

        const int c = ntile * 16 + m;   // weight column 0..255
        const int j = ntile >> 2;       // compile-time constant after unroll
        const int u = c & 63;           // column within the 64-block

        #pragma unroll
        for (int r = 0; r < 8; ++r) {
            const int    el = erow0 + hi * 8 + r;           // local row (D layout)
            const size_t e  = (size_t)(eblock + el);        // global row
            const float  w  = acc[r];
            const float* x2 = &lds_x2[el * 4];
            const float* d1 = &data1[e * 256];
            float*       o  = &out[e * 512];
            if (j == 0) {                                   // out_k0: cols 0..63
                __builtin_nontemporal_store(w * d1[u] * x2[0], &o[u]);
            } else if (j == 1) {                            // out_k1: cols 128..319
                float tw = w * d1[u];
                __builtin_nontemporal_store(tw * x2[1], &o[128 + 3 * u + 0]);
                __builtin_nontemporal_store(tw * x2[2], &o[128 + 3 * u + 1]);
                __builtin_nontemporal_store(tw * x2[3], &o[128 + 3 * u + 2]);
            } else if (j == 2) {                            // out_k2: cols 320..511
                float tw = w * x2[0];
                __builtin_nontemporal_store(tw * d1[64 + 3 * u + 0], &o[320 + 3 * u + 0]);
                __builtin_nontemporal_store(tw * d1[64 + 3 * u + 1], &o[320 + 3 * u + 1]);
                __builtin_nontemporal_store(tw * d1[64 + 3 * u + 2], &o[320 + 3 * u + 2]);
            } else {                                        // out_k3: cols 64..127
                float dotv = d1[64 + 3 * u + 0] * x2[1]
                           + d1[64 + 3 * u + 1] * x2[2]
                           + d1[64 + 3 * u + 2] * x2[3];
                __builtin_nontemporal_store(w * dotv * INV_SQRT3_F, &o[64 + u]);
            }
        }
    }
}

extern "C" void kernel_launch(void* const* d_in, const int* in_sizes, int n_in,
                              void* d_out, int out_size, void* d_ws, size_t ws_size,
                              hipStream_t stream) {
    const float* data1   = (const float*)d_in[0];
    const float* data2   = (const float*)d_in[1];
    const float* dweight = (const float*)d_in[2];
    const float* w0      = (const float*)d_in[3];
    const float* w1      = (const float*)d_in[4];
    float* out = (float*)d_out;

    const int E      = in_sizes[2] / 8;            // 300000
    const int blocks = E / ROWS_PER_BLOCK;         // 300000 / 96 = 3125 exactly

    UVUTensorProduct_17420387352803_kernel<<<blocks, THREADS, 0, stream>>>(
        data1, data2, dweight, w0, w1, out);
}
